// HybridPolicy_9715216023865
// MI455X (gfx1250) — matrix operand order repaired and checked by reading, not compiled, not run on hardware
//
#include <hip/hip_runtime.h>
#include <hip/hip_bf16.h>

typedef __attribute__((ext_vector_type(16))) _Float16 v16h;
typedef __attribute__((ext_vector_type(8)))  float    v8f;

// Problem constants (match reference)
#define C_TOT   128000
#define C_B     128
#define C_E     512000
#define C_HEADS 4
#define C_H     64
#define C_FIN   32
#define C_A     15
#define C_OBS   1500
#define C_HS    512
#define C_NOUT  15
#define C_VH    128
#define C_L     4

// ---------------------------------------------------------------------------
// CDNA5 async global->LDS staging (builtin confirmed present by round-2
// diagnostic: takes AS1 int* src, AS3 int* dst, imm offset, imm cpol).
// Falls back to synchronous copy when the toolchain doesn't declare it.
// ---------------------------------------------------------------------------
#if defined(__gfx1250__) && __has_builtin(__builtin_amdgcn_global_load_async_to_lds_b32)
#define HAVE_ASYNC_LDS 1
#endif

__device__ __forceinline__ void stage_to_lds(const float* __restrict__ g,
                                             float* l, int count) {
  const int tid = threadIdx.x;
  const int nthr = blockDim.x;
#ifdef HAVE_ASYNC_LDS
  typedef __attribute__((address_space(1))) int GInt;
  typedef __attribute__((address_space(3))) int LInt;
  for (int i = tid; i < count; i += nthr)
    __builtin_amdgcn_global_load_async_to_lds_b32((GInt*)(g + i),
                                                  (LInt*)(l + i), 0, 0);
#if __has_builtin(__builtin_amdgcn_s_wait_asynccnt)
  __builtin_amdgcn_s_wait_asynccnt(0);
#else
  asm volatile("s_wait_asynccnt 0" ::: "memory");
#endif
#else
  for (int i = tid; i < count; i += nthr) l[i] = g[i];
#endif
}

// ---------------------------------------------------------------------------
// WMMA tile helpers.
// A-layout (16-bit A 16x32): lanes 0-15 row M=lane, K = kb+0..15;
//                            lanes 16-31 row M=lane-16, K = kb+16..31.
// B-layout (16-bit B 32x16): lanes 0-15 col N=lane, K = kb+0..15;
//                            lanes 16-31 col N=lane-16, K = kb+16..31.
// C/D layout: VGPR r -> (M = r + (lane<16?0:8), N = lane&15).
// ---------------------------------------------------------------------------

// Generic WMMA GEMM: C[M,ND] = act(A[M,KD] @ W[KD,ND] + bias)
// Block = (ND/16) waves; block covers 16 rows; weights f32-staged in LDS.
template <int KD, int ND, int ACT>  // ACT: 0=none, 1=relu
__global__ void k_wmma_gemm(const float* __restrict__ A,
                            const float* __restrict__ W,
                            const float* __restrict__ bias,
                            float* __restrict__ C, int M) {
  __shared__ float sW[KD * ND];
  stage_to_lds(W, sW, KD * ND);
  __syncthreads();

  const int lane = threadIdx.x & 31;
  const int wid  = threadIdx.x >> 5;
  const int m0   = blockIdx.x * 16;
  const int n0   = wid * 16;
  const int hi   = lane >> 4;
  const int row  = m0 + (lane & 15);
  const int col  = lane & 15;
  const int aklo = hi * 8;
  const int bkof = hi * 16;

  v8f acc = {};
  const float* arow = A + (size_t)row * KD;
#pragma unroll
  for (int kb = 0; kb < KD; kb += 32) {
    v16h a, b;
#pragma unroll
    for (int j = 0; j < 16; ++j)
      a[j] = (_Float16)arow[kb + bkof + j];
#pragma unroll
    for (int j = 0; j < 16; ++j)
      b[j] = (_Float16)sW[(size_t)(kb + bkof + j) * ND + n0 + col];
    acc = __builtin_amdgcn_wmma_f32_16x16x32_f16(false, a, false, b, (short)0,
                                                 acc, false, false);
  }

  const int orow = m0 + hi * 8;
  const float bj = bias ? bias[n0 + col] : 0.0f;
#pragma unroll
  for (int r = 0; r < 8; ++r) {
    float vv = acc[r] + bj;
    if (ACT == 1) vv = fmaxf(vv, 0.0f);
    C[(size_t)(orow + r) * ND + n0 + col] = vv;
  }
}

// Fused Q/K/V projection: reads each A-tile once, 3 WMMAs per K-step.
__global__ void k_wmma_qkv(const float* __restrict__ A,
                           const float* __restrict__ Wq,
                           const float* __restrict__ Wk,
                           const float* __restrict__ Wv,
                           float* __restrict__ Q, float* __restrict__ Kx,
                           float* __restrict__ V, int M) {
  __shared__ float sW[3 * C_H * C_H];
  stage_to_lds(Wq, sW, C_H * C_H);
  stage_to_lds(Wk, sW + C_H * C_H, C_H * C_H);
  stage_to_lds(Wv, sW + 2 * C_H * C_H, C_H * C_H);
  __syncthreads();

  const int lane = threadIdx.x & 31;
  const int wid  = threadIdx.x >> 5;   // 4 waves, one 16-col tile each
  const int m0   = blockIdx.x * 16;
  const int n0   = wid * 16;
  const int hi   = lane >> 4;
  const int row  = m0 + (lane & 15);
  const int col  = lane & 15;
  const int aklo = hi * 8;
  const int bkof = hi * 16;

  v8f aq = {}, ak = {}, av = {};
  const float* arow = A + (size_t)row * C_H;
#pragma unroll
  for (int kb = 0; kb < C_H; kb += 32) {
    v16h a, bq, bk, bv;
#pragma unroll
    for (int j = 0; j < 16; ++j)
      a[j] = (_Float16)arow[kb + bkof + j];
#pragma unroll
    for (int j = 0; j < 16; ++j) {
      const int widx = (kb + bkof + j) * C_H + n0 + col;
      bq[j] = (_Float16)sW[widx];
      bk[j] = (_Float16)sW[C_H * C_H + widx];
      bv[j] = (_Float16)sW[2 * C_H * C_H + widx];
    }
    aq = __builtin_amdgcn_wmma_f32_16x16x32_f16(false, a, false, bq, (short)0,
                                                aq, false, false);
    ak = __builtin_amdgcn_wmma_f32_16x16x32_f16(false, a, false, bk, (short)0,
                                                ak, false, false);
    av = __builtin_amdgcn_wmma_f32_16x16x32_f16(false, a, false, bv, (short)0,
                                                av, false, false);
  }

  const int orow = m0 + hi * 8;
#pragma unroll
  for (int r = 0; r < 8; ++r) {
    const size_t o = (size_t)(orow + r) * C_H + n0 + col;
    Q[o] = aq[r];
    Kx[o] = ak[r];
    V[o] = av[r];
  }
}

// Fused softmax-normalize + output projection:
// C = ((agg / (denom+1e-9)) @ Wo); normalization folded into the A-load.
__global__ void k_wmma_attn_o(const float* __restrict__ agg,
                              const float* __restrict__ denom,
                              const float* __restrict__ W,
                              float* __restrict__ C, int M) {
  __shared__ float sW[C_H * C_H];
  stage_to_lds(W, sW, C_H * C_H);
  __syncthreads();

  const int lane = threadIdx.x & 31;
  const int wid  = threadIdx.x >> 5;
  const int m0   = blockIdx.x * 16;
  const int n0   = wid * 16;
  const int hi   = lane >> 4;
  const int row  = m0 + (lane & 15);
  const int col  = lane & 15;
  const int aklo = hi * 8;
  const int bkof = hi * 16;

  float rd[C_HEADS];
#pragma unroll
  for (int h = 0; h < C_HEADS; ++h)
    rd[h] = 1.0f / (denom[(size_t)row * C_HEADS + h] + 1e-9f);

  v8f acc = {};
  const float* arow = agg + (size_t)row * C_H;
#pragma unroll
  for (int kb = 0; kb < C_H; kb += 32) {
    v16h a, b;
#pragma unroll
    for (int j = 0; j < 16; ++j) {
      const int koff = kb + bkof + j;
      a[j] = (_Float16)(arow[koff] * rd[koff >> 4]);
    }
#pragma unroll
    for (int j = 0; j < 16; ++j)
      b[j] = (_Float16)sW[(size_t)(kb + bkof + j) * C_H + n0 + col];
    acc = __builtin_amdgcn_wmma_f32_16x16x32_f16(false, a, false, b, (short)0,
                                                 acc, false, false);
  }

  const int orow = m0 + hi * 8;
#pragma unroll
  for (int r = 0; r < 8; ++r)
    C[(size_t)(orow + r) * C_H + n0 + col] = acc[r];
}

// ---------------------------------------------------------------------------
// Attention state init: segment-max keys to -inf-key, denom/agg to zero.
// ---------------------------------------------------------------------------
__global__ void k_init_attn(int* __restrict__ mI, float* __restrict__ denom,
                            float* __restrict__ agg, int n4, int n64) {
  int t = blockIdx.x * blockDim.x + threadIdx.x;
  if (t < n4) {
    mI[t] = (int)0x80000000u;  // key-space -inf
    denom[t] = 0.0f;
  }
  if (t < n64) agg[t] = 0.0f;
}

// ---------------------------------------------------------------------------
// Per-(edge,head) attention scores + ordered-int atomic segment max.
// ---------------------------------------------------------------------------
__global__ void k_edge_score(const float* __restrict__ q,
                             const float* __restrict__ k,
                             const int* __restrict__ src,
                             const int* __restrict__ dst,
                             float* __restrict__ s, int* __restrict__ mI) {
  int t = blockIdx.x * blockDim.x + threadIdx.x;
  if (t >= C_E * C_HEADS) return;
  int e = t >> 2;
  int h = t & 3;
  if (e + 2048 < C_E) {
    __builtin_prefetch(src + e + 2048, 0, 0);  // -> global_prefetch_b8
    __builtin_prefetch(dst + e + 2048, 0, 0);
  }
  int sn = src[e], dn = dst[e];
  const float* qd = q + (size_t)dn * C_H + h * 16;
  const float* ks = k + (size_t)sn * C_H + h * 16;
  float acc = 0.0f;
#pragma unroll
  for (int i = 0; i < 16; ++i) acc += qd[i] * ks[i];
  acc *= 0.25f;  // 1/sqrt(DH), DH=16
  s[t] = acc;
  int bits = __float_as_int(acc);
  int key = (bits < 0) ? (bits ^ 0x7fffffff) : bits;  // order-preserving map
  atomicMax(mI + dn * C_HEADS + h, key);
}

// ---------------------------------------------------------------------------
// Per-(edge,head) softmax weight + atomic segment sums (denom, weighted V).
// ---------------------------------------------------------------------------
__global__ void k_edge_accum(const float* __restrict__ v,
                             const int* __restrict__ src,
                             const int* __restrict__ dst,
                             const float* __restrict__ s,
                             const int* __restrict__ mI,
                             float* __restrict__ denom,
                             float* __restrict__ agg) {
  int t = blockIdx.x * blockDim.x + threadIdx.x;
  if (t >= C_E * C_HEADS) return;
  int e = t >> 2;
  int h = t & 3;
  int sn = src[e], dn = dst[e];
  int key = mI[dn * C_HEADS + h];
  int bits = (key < 0) ? (key ^ 0x7fffffff) : key;
  float m = __int_as_float(bits);
  float w = __expf(s[t] - m);
  atomicAdd(denom + dn * C_HEADS + h, w);
  const float* vs = v + (size_t)sn * C_H + h * 16;
  float* ag = agg + (size_t)dn * C_H + h * 16;
#pragma unroll
  for (int i = 0; i < 16; ++i) atomicAdd(ag + i, w * vs[i]);
}

// ---------------------------------------------------------------------------
// Fused residual + LayerNorm over H=64: one wave per row (wave32, 2 el/lane).
// x = LN(x + y) * g + b
// ---------------------------------------------------------------------------
__global__ void k_add_ln(float* __restrict__ x, const float* __restrict__ y,
                         const float* __restrict__ g,
                         const float* __restrict__ b, int M) {
  int row = blockIdx.x * (blockDim.x >> 5) + (threadIdx.x >> 5);
  if (row >= M) return;
  int lane = threadIdx.x & 31;
  size_t base = (size_t)row * C_H;
  float v0 = x[base + lane] + y[base + lane];
  float v1 = x[base + 32 + lane] + y[base + 32 + lane];
  float sum = v0 + v1;
#pragma unroll
  for (int off = 16; off > 0; off >>= 1) sum += __shfl_xor(sum, off, 32);
  float mean = sum * (1.0f / 64.0f);
  float d0 = v0 - mean, d1 = v1 - mean;
  float var = d0 * d0 + d1 * d1;
#pragma unroll
  for (int off = 16; off > 0; off >>= 1) var += __shfl_xor(var, off, 32);
  var *= (1.0f / 64.0f);
  float inv = rsqrtf(var + 1e-5f);
  x[base + lane]      = d0 * inv * g[lane]      + b[lane];
  x[base + 32 + lane] = d1 * inv * g[32 + lane] + b[32 + lane];
}

// ---------------------------------------------------------------------------
// Readout: gat_out[b,a] = x[agent[b]] . Wr[:,a] + br[a]
// ---------------------------------------------------------------------------
__global__ void k_readout(const float* __restrict__ x,
                          const int* __restrict__ agent,
                          const float* __restrict__ Wr,
                          const float* __restrict__ br,
                          float* __restrict__ out) {
  int t = blockIdx.x * blockDim.x + threadIdx.x;
  if (t >= C_B * C_A) return;
  int bb = t / C_A, aa = t % C_A;
  const float* xr = x + (size_t)agent[bb] * C_H;
  float acc = br[aa];
#pragma unroll 8
  for (int i = 0; i < C_H; ++i) acc += xr[i] * Wr[i * C_A + aa];
  out[t] = acc;
}

// feat = tanh(concat(gat_out[b,:A], obs[b,:OBS]) @ Wp1 + bp1)
__global__ void k_fc_concat(const float* __restrict__ gat_out,
                            const float* __restrict__ obs,
                            const float* __restrict__ W,
                            const float* __restrict__ bias,
                            float* __restrict__ out) {
  int t = blockIdx.x * blockDim.x + threadIdx.x;
  if (t >= C_B * C_HS) return;
  int bb = t / C_HS, j = t % C_HS;
  float acc = bias[j];
  const float* g = gat_out + bb * C_A;
#pragma unroll
  for (int i = 0; i < C_A; ++i) acc += g[i] * W[i * C_HS + j];
  const float* o = obs + (size_t)bb * C_OBS;
  const float* Wo_ = W + C_A * C_HS;
  for (int i = 0; i < C_OBS; ++i) acc += o[i] * Wo_[i * C_HS + j];
  out[t] = tanhf(acc);
}

// generic small FC: out[M,ND] = act(in[M,KD] @ W + b); act 0=none, 2=tanh
__global__ void k_fc(const float* __restrict__ in, const float* __restrict__ W,
                     const float* __restrict__ bias, float* __restrict__ out,
                     int M, int KD, int ND, int act) {
  int t = blockIdx.x * blockDim.x + threadIdx.x;
  if (t >= M * ND) return;
  int r = t / ND, j = t % ND;
  float acc = bias[j];
  const float* ir = in + (size_t)r * KD;
  for (int i = 0; i < KD; ++i) acc += ir[i] * W[i * ND + j];
  out[t] = (act == 2) ? tanhf(acc) : acc;
}

// ---------------------------------------------------------------------------
extern "C" void kernel_launch(void* const* d_in, const int* in_sizes, int n_in,
                              void* d_out, int out_size, void* d_ws,
                              size_t ws_size, hipStream_t stream) {
  const float* node_feats = (const float*)d_in[0];
  const float* obs        = (const float*)d_in[1];
  const int*   esrc       = (const int*)d_in[2];
  const int*   edst       = (const int*)d_in[3];
  const int*   agents     = (const int*)d_in[4];
  const float* W_in  = (const float*)d_in[5];
  const float* Wq    = (const float*)d_in[6];
  const float* Wk    = (const float*)d_in[7];
  const float* Wv    = (const float*)d_in[8];
  const float* Wo    = (const float*)d_in[9];
  const float* ln1_g = (const float*)d_in[10];
  const float* ln1_b = (const float*)d_in[11];
  const float* W1    = (const float*)d_in[12];
  const float* b1    = (const float*)d_in[13];
  const float* W2    = (const float*)d_in[14];
  const float* b2    = (const float*)d_in[15];
  const float* ln2_g = (const float*)d_in[16];
  const float* ln2_b = (const float*)d_in[17];
  const float* Wr    = (const float*)d_in[18];
  const float* br    = (const float*)d_in[19];
  const float* Wp1   = (const float*)d_in[20];
  const float* bp1   = (const float*)d_in[21];
  const float* Wp2   = (const float*)d_in[22];
  const float* bp2   = (const float*)d_in[23];
  const float* Wlog  = (const float*)d_in[24];
  const float* blog  = (const float*)d_in[25];
  const float* Wv1   = (const float*)d_in[26];
  const float* bv1   = (const float*)d_in[27];
  const float* Wv2   = (const float*)d_in[28];
  const float* bv2   = (const float*)d_in[29];
  const float* Wvo   = (const float*)d_in[30];
  const float* bvo   = (const float*)d_in[31];

  // Workspace layout (floats)
  float* ws    = (float*)d_ws;
  float* x     = ws;                                // TOT*64
  float* t1    = x    + (size_t)C_TOT * C_H;        // TOT*64 (q / ffn-out)
  float* t2    = t1   + (size_t)C_TOT * C_H;        // TOT*64 (k / attn@Wo)
  float* t3    = t2   + (size_t)C_TOT * C_H;        // TOT*64 (v)
  float* agg   = t3   + (size_t)C_TOT * C_H;        // TOT*64
  float* hbuf  = agg  + (size_t)C_TOT * C_H;        // TOT*128 (ffn hidden / s[E*4])
  float* denom = hbuf + (size_t)C_TOT * 2 * C_H;    // TOT*4
  int*   mI    = (int*)(denom + (size_t)C_TOT * C_HEADS);  // TOT*4
  float* gat_out = (float*)(mI + (size_t)C_TOT * C_HEADS); // B*15
  float* feat    = gat_out + C_B * C_A;             // B*512
  float* feat2   = feat + C_B * C_HS;               // B*512
  float* vh      = feat2 + C_B * C_HS;              // B*128
  float* vh2     = vh + C_B * C_VH;                 // B*128
  float* sbuf    = hbuf;                            // E*HEADS scores overlay

  const int rowBlocks = C_TOT / 16;                 // 8000
  const int eThreads  = C_E * C_HEADS;              // 2.05M
  const int eBlocks   = (eThreads + 255) / 256;
  const int nElem     = C_TOT * C_H;
  const int nBlocks   = (nElem + 255) / 256;
  const int lnBlocks  = C_TOT / 8;                  // 8 rows/block (256 thr)

  // Input projection: x = node_feats @ W_in   [TOT,32]x[32,64]
  k_wmma_gemm<C_FIN, C_H, 0><<<rowBlocks, 128, 0, stream>>>(
      node_feats, W_in, nullptr, x, C_TOT);

  for (int l = 0; l < C_L; ++l) {
    const size_t wHH = (size_t)l * C_H * C_H;
    // fused q, k, v (A-tile read once, 3 WMMAs per K-step)
    k_wmma_qkv<<<rowBlocks, 128, 0, stream>>>(x, Wq + wHH, Wk + wHH, Wv + wHH,
                                              t1, t2, t3, C_TOT);
    // segment softmax over incoming edges
    k_init_attn<<<nBlocks, 256, 0, stream>>>(mI, denom, agg,
                                             C_TOT * C_HEADS, nElem);
    k_edge_score<<<eBlocks, 256, 0, stream>>>(t1, t2, esrc, edst, sbuf, mI);
    k_edge_accum<<<eBlocks, 256, 0, stream>>>(t3, esrc, edst, sbuf, mI,
                                              denom, agg);
    // fused normalize + output projection, then residual LN
    k_wmma_attn_o<<<rowBlocks, 128, 0, stream>>>(agg, denom, Wo + wHH, t2,
                                                 C_TOT);
    k_add_ln<<<lnBlocks, 256, 0, stream>>>(x, t2, ln1_g + l * C_H,
                                           ln1_b + l * C_H, C_TOT);
    // FFN
    k_wmma_gemm<C_H, 2 * C_H, 1><<<rowBlocks, 256, 0, stream>>>(
        x, W1 + (size_t)l * C_H * 2 * C_H, b1 + l * 2 * C_H, hbuf, C_TOT);
    k_wmma_gemm<2 * C_H, C_H, 0><<<rowBlocks, 128, 0, stream>>>(
        hbuf, W2 + (size_t)l * 2 * C_H * C_H, b2 + l * C_H, t1, C_TOT);
    k_add_ln<<<lnBlocks, 256, 0, stream>>>(x, t1, ln2_g + l * C_H,
                                           ln2_b + l * C_H, C_TOT);
  }

  // Heads
  k_readout<<<(C_B * C_A + 255) / 256, 256, 0, stream>>>(x, agents, Wr, br,
                                                         gat_out);
  k_fc_concat<<<(C_B * C_HS + 255) / 256, 256, 0, stream>>>(gat_out, obs, Wp1,
                                                            bp1, feat);
  k_fc<<<(C_B * C_HS + 255) / 256, 256, 0, stream>>>(feat, Wp2, bp2, feat2,
                                                     C_B, C_HS, C_HS, 2);
  k_fc<<<(C_B * C_NOUT + 255) / 256, 256, 0, stream>>>(
      feat2, Wlog, blog, (float*)d_out, C_B, C_HS, C_NOUT, 0);
  k_fc<<<(C_B * C_VH + 255) / 256, 256, 0, stream>>>(obs, Wv1, bv1, vh, C_B,
                                                     C_OBS, C_VH, 2);
  k_fc<<<(C_B * C_VH + 255) / 256, 256, 0, stream>>>(vh, Wv2, bv2, vh2, C_B,
                                                     C_VH, C_VH, 2);
  k_fc<<<(C_B + 255) / 256, 256, 0, stream>>>(
      vh2, Wvo, bvo, (float*)d_out + C_B * C_NOUT, C_B, C_VH, 1, 0);
}
